// Decoder_63505386438998
// MI455X (gfx1250) — matrix operand order repaired and checked by reading, not compiled, not run on hardware
//
#include <hip/hip_runtime.h>
#include <hip/hip_bf16.h>
#include <math.h>

// Problem constants (from reference)
#define B_   16
#define S_   256
#define D_   768
#define H_   12
#define L_   6
#define DFF_ 2048
#define HD_  64
#define IMG_ 256
#define CPP_ 768   // C*P*P == D

// WMMA tile config: 256 threads = 8 wave32; block tile 64x128, K-step 32.
#define BM 64
#define BN 128
#define BK 32

typedef __attribute__((ext_vector_type(16))) __bf16 v16bf;
typedef __attribute__((ext_vector_type(8)))  float  v8f;

union FragAB { unsigned u[8]; v16bf v; };

// ---------------------------------------------------------------------------
// CDNA5 async-copy helpers (GLOBAL_LOAD_ASYNC_TO_LDS_*, ASYNCcnt; ISA ch.10/15.18)
// ---------------------------------------------------------------------------
__device__ __forceinline__ unsigned lds_lo32(const void* p)
{
    // Flat->LDS translation uses addr[31:0] as the wave-relative LDS address.
    return (unsigned)(unsigned long long)(uintptr_t)p;
}

__device__ __forceinline__ void async_copy_b128(unsigned lds_addr, const void* gptr)
{
    unsigned long long ga = (unsigned long long)(uintptr_t)gptr;
    asm volatile("global_load_async_to_lds_b128 %0, %1, off"
                 :: "v"(lds_addr), "v"(ga) : "memory");
}

__device__ __forceinline__ void wait_asynccnt0()
{
    asm volatile("s_wait_asynccnt 0x0" ::: "memory");
}

// ---------------------------------------------------------------------------
// GEMM (bf16 in, fp32 accum):  C[m,n] = alpha * sum_k A[m,k] * B(n,k) (+bias)(+ReLU)
//   TRANSB=true : B element (n,k) at n*ldb + k  (weights / K^T) -> async B staging
//   TRANSB=false: B element (k,n) at k*ldb + n  (attn @ V)      -> sync transposing B staging
// A staging is always async. Double-buffered LDS, one barrier per K-step:
//   wait(asynccnt)=tile i ready -> barrier -> issue tile i+1 -> WMMA on tile i.
// Optional fp32 and/or bf16 outputs (Cbf feeds the next GEMM directly).
// Batched over gridDim.z; z -> (b=z/Hh, h=z%Hh) with per-b/per-h element strides.
// ---------------------------------------------------------------------------
template<bool TRANSB>
__global__ __launch_bounds__(256)
void gemm_bf16_wmma(const __bf16* __restrict__ A, size_t sAb, size_t sAh, int lda,
                    const __bf16* __restrict__ Bm, size_t sBb, size_t sBh, int ldb,
                    const float* __restrict__ bias,
                    float* __restrict__ C, __bf16* __restrict__ Cbf,
                    size_t sCb, size_t sCh, int ldc,
                    int M, int N, int K, float alpha, int relu, int Hh)
{
    __shared__ __bf16 aS[2][BM * BK];   // [m][k]
    __shared__ __bf16 bS[2][BN * BK];   // [n][k]

    const int z  = blockIdx.z;
    const int bb = z / Hh;
    const int hh = z % Hh;
    const __bf16* Ab = A  + (size_t)bb * sAb + (size_t)hh * sAh;
    const __bf16* Bb = Bm + (size_t)bb * sBb + (size_t)hh * sBh;

    const int m0 = blockIdx.y * BM;
    const int n0 = blockIdx.x * BN;

    const int t    = threadIdx.x;
    const int lane = t & 31;
    const int w    = t >> 5;
    const int wm   = w >> 2;    // 0..1  -> 32-row group
    const int wn   = w & 3;     // 0..3  -> 32-col group

    // --- per-thread staging coordinates -----------------------------------
    // A tile: 64x32 bf16 = 4KB = 256 lanes x 16B (one b128 each)
    const int ar = t >> 2;            // row in tile
    const int ak = (t & 3) * 8;       // k in tile
    int agm = m0 + ar; if (agm > M - 1) agm = M - 1;
    const __bf16* agp = Ab + (size_t)agm * lda + ak;
    // B tile (TRANSB): 128x32 bf16 = 8KB = 256 lanes x 32B (two b128 each)
    const int br = t >> 1;            // n in tile
    const int bk = (t & 1) * 16;      // k in tile (0 or 16)
    int bgn = n0 + br; if (bgn > N - 1) bgn = N - 1;
    const __bf16* bgp = Bb + (size_t)bgn * ldb + bk;

    v8f acc[2][2];
#pragma unroll
    for (int i = 0; i < 2; ++i)
#pragma unroll
        for (int j = 0; j < 2; ++j)
#pragma unroll
            for (int r = 0; r < 8; ++r) acc[i][j][r] = 0.0f;

    auto stage = [&](int k0, int buf) {
        async_copy_b128(lds_lo32(&aS[buf][ar * BK + ak]), agp + k0);
        if (TRANSB) {
            async_copy_b128(lds_lo32(&bS[buf][br * BK + bk]), bgp + k0);
            async_copy_b128(lds_lo32(&bS[buf][br * BK + bk + 8]), bgp + k0 + 8);
        } else {
            // transpose while staging: B[k][n] -> LDS [n][k]
#pragma unroll
            for (int i = 0; i < 16; ++i) {
                const int e  = t + i * 256;
                const int kk = e >> 7;    // k in tile
                const int nn = e & 127;   // n in tile
                int gn = n0 + nn; if (gn > N - 1) gn = N - 1;
                bS[buf][nn * BK + kk] = Bb[(size_t)(k0 + kk) * ldb + gn];
            }
        }
    };

    auto compute = [&](int buf) {
        const __bf16* a = aS[buf];
        const __bf16* b = bS[buf];
        FragAB fa[2], fb[2];
        const int l15 = lane & 15;
        const int hiA = ((lane >> 4) & 1) << 2;  // +4 pair-groups for lanes 16-31
        const int hiB = ((lane >> 4) & 1) << 2;  // +4 pair-groups for lanes 16-31
#pragma unroll
        for (int i = 0; i < 2; ++i) {
            const int m = wm * 32 + i * 16 + l15;
            const unsigned* row = (const unsigned*)(a + m * BK); // 16 K-pairs
#pragma unroll
            for (int vv = 0; vv < 8; ++vv) {
                const int p = (vv < 4 ? vv : vv + 4) + hiA;
                fa[i].u[vv] = row[p];
            }
        }
#pragma unroll
        for (int j = 0; j < 2; ++j) {
            const int n = wn * 32 + j * 16 + l15;
            const unsigned* row = (const unsigned*)(b + n * BK);
#pragma unroll
            for (int vv = 0; vv < 8; ++vv) fb[j].u[vv] = row[(vv < 4 ? vv : vv + 4) + hiB];
        }
#pragma unroll
        for (int i = 0; i < 2; ++i)
#pragma unroll
            for (int j = 0; j < 2; ++j)
                acc[i][j] = __builtin_amdgcn_wmma_f32_16x16x32_bf16(
                    false, fa[i].v, false, fb[j].v,
                    (short)0, acc[i][j], false, false);
    };

    // --- double-buffered pipeline ----------------------------------------
    const int nk = K / BK;   // all K are multiples of 32
    stage(0, 0);
    for (int i = 0; i < nk; ++i) {
        wait_asynccnt0();           // this wave's loads for tile i complete
        __syncthreads();            // whole block's tile i visible; prev compute done
        if (i + 1 < nk) stage((i + 1) * BK, (i + 1) & 1);
        compute(i & 1);             // overlaps with async fill of the other buffer
    }

    // --- epilogue: alpha, bias, relu, dual fp32/bf16 store ----------------
    float*  Cf = C   ? (C   + (size_t)bb * sCb + (size_t)hh * sCh) : nullptr;
    __bf16* Ch = Cbf ? (Cbf + (size_t)bb * sCb + (size_t)hh * sCh) : nullptr;
    const int hiM = ((lane >> 4) & 1) * 8;
    const int l15 = lane & 15;
#pragma unroll
    for (int i = 0; i < 2; ++i) {
#pragma unroll
        for (int j = 0; j < 2; ++j) {
            const int ncol = n0 + wn * 32 + j * 16 + l15;
            const float bval = (bias && ncol < N) ? bias[ncol] : 0.0f;
#pragma unroll
            for (int r = 0; r < 8; ++r) {
                const int mrow = m0 + wm * 32 + i * 16 + r + hiM;
                if (mrow < M && ncol < N) {
                    float val = acc[i][j][r] * alpha + bval;
                    if (relu) val = fmaxf(val, 0.0f);
                    const size_t idx = (size_t)mrow * ldc + ncol;
                    if (Cf) Cf[idx] = val;
                    if (Ch) Ch[idx] = (__bf16)val;
                }
            }
        }
    }
}

// ---------------------------------------------------------------------------
// Row softmax over S=256 cols; writes bf16 probabilities for the AV GEMM.
// ---------------------------------------------------------------------------
__global__ __launch_bounds__(256)
void softmax_rows(const float* __restrict__ scores, __bf16* __restrict__ probs,
                  int causal)
{
    const size_t row = blockIdx.x;
    const int q = (int)(row % S_);
    const float* p = scores + row * S_;
    __bf16* o = probs + row * S_;
    const int t = threadIdx.x;
    __shared__ float red[256];

    float v = p[t];
    const bool masked = causal && (t > q);
    if (masked) v = -INFINITY;

    red[t] = v; __syncthreads();
#pragma unroll
    for (int s = 128; s > 0; s >>= 1) {
        if (t < s) red[t] = fmaxf(red[t], red[t + s]);
        __syncthreads();
    }
    const float mx = red[0];
    __syncthreads();

    const float e = masked ? 0.0f : __expf(v - mx);
    red[t] = e; __syncthreads();
#pragma unroll
    for (int s = 128; s > 0; s >>= 1) {
        if (t < s) red[t] += red[t + s];
        __syncthreads();
    }
    o[t] = (__bf16)(e / red[0]);
}

// ---------------------------------------------------------------------------
// out = LayerNorm(x + r) * g + b ; also writes bf16 mirror (next GEMM's A).
// Safe when out == x.
// ---------------------------------------------------------------------------
__global__ __launch_bounds__(256)
void add_layernorm(const float* __restrict__ x, const float* __restrict__ r,
                   const float* __restrict__ g, const float* __restrict__ b,
                   float* __restrict__ out, __bf16* __restrict__ outbf, int D)
{
    const size_t row = blockIdx.x;
    const float* xr = x + row * (size_t)D;
    const float* rr = r + row * (size_t)D;
    float* orow = out + row * (size_t)D;
    __bf16* obf = outbf + row * (size_t)D;
    const int t = threadIdx.x;
    __shared__ float s1[256], s2[256];

    float lsum = 0.0f, lsq = 0.0f;
    for (int c = t; c < D; c += 256) {
        const float v = xr[c] + rr[c];
        lsum += v; lsq += v * v;
    }
    s1[t] = lsum; s2[t] = lsq; __syncthreads();
#pragma unroll
    for (int s = 128; s > 0; s >>= 1) {
        if (t < s) { s1[t] += s1[t + s]; s2[t] += s2[t + s]; }
        __syncthreads();
    }
    const float mean = s1[0] / (float)D;
    const float var  = s2[0] / (float)D - mean * mean;
    const float rstd = rsqrtf(var + 1e-5f);

    for (int c = t; c < D; c += 256) {
        const float v = (xr[c] + rr[c] - mean) * rstd * g[c] + b[c];
        orow[c] = v;
        obf[c] = (__bf16)v;
    }
}

// ---------------------------------------------------------------------------
// x = encoded + sinusoidal PE  (fp32 + bf16 mirrors)
// ---------------------------------------------------------------------------
__global__ __launch_bounds__(256)
void add_pe(const float* __restrict__ in, float* __restrict__ out,
            __bf16* __restrict__ outbf, int total)
{
    const int idx = blockIdx.x * 256 + threadIdx.x;
    if (idx >= total) return;
    const int c = idx % D_;
    const int s = (idx / D_) % S_;
    const int i2 = c & ~1;  // 2*i
    const float ang = (float)s * __expf(-(float)i2 * (9.210340371976184f / (float)D_));
    const float pe = (c & 1) ? __cosf(ang) : __sinf(ang);
    const float v = in[idx] + pe;
    out[idx] = v;
    outbf[idx] = (__bf16)v;
}

// ---------------------------------------------------------------------------
// fp32 -> bf16 bulk convert (weights, once per call)
// ---------------------------------------------------------------------------
__global__ __launch_bounds__(256)
void f32_to_bf16(const float* __restrict__ in, __bf16* __restrict__ out, int n)
{
    const int idx = (blockIdx.x * 256 + threadIdx.x) * 4;
    if (idx + 3 < n) {
        const float4 v = *(const float4*)(in + idx);
        out[idx + 0] = (__bf16)v.x;
        out[idx + 1] = (__bf16)v.y;
        out[idx + 2] = (__bf16)v.z;
        out[idx + 3] = (__bf16)v.w;
    } else {
        for (int i = idx; i < n; ++i) out[i] = (__bf16)in[i];
    }
}

// ---------------------------------------------------------------------------
// dst[b][c][i][j] = src[((b*IMG + i)*IMG + j)*3 + c]   (NHWC view -> NCHW)
// ---------------------------------------------------------------------------
__global__ __launch_bounds__(256)
void out_transpose(const float* __restrict__ src, float* __restrict__ dst, int total)
{
    const int idx = blockIdx.x * 256 + threadIdx.x;
    if (idx >= total) return;
    const int j = idx % IMG_;
    const int i = (idx / IMG_) % IMG_;
    const int c = (idx / (IMG_ * IMG_)) % 3;
    const int b = idx / (3 * IMG_ * IMG_);
    dst[idx] = src[((size_t)(b * IMG_ + i) * IMG_ + j) * 3 + c];
}

// ---------------------------------------------------------------------------
// Host-side launch helpers
// ---------------------------------------------------------------------------
static inline void launch_gemm(hipStream_t stream, bool transB,
                               const __bf16* A, size_t sAb, size_t sAh, int lda,
                               const __bf16* Bm, size_t sBb, size_t sBh, int ldb,
                               const float* bias,
                               float* C, __bf16* Cbf, size_t sCb, size_t sCh, int ldc,
                               int M, int N, int K, float alpha, int relu,
                               int Hh, int batch)
{
    dim3 grid((N + BN - 1) / BN, (M + BM - 1) / BM, batch);
    if (transB)
        gemm_bf16_wmma<true><<<grid, 256, 0, stream>>>(A, sAb, sAh, lda, Bm, sBb, sBh, ldb,
                                                       bias, C, Cbf, sCb, sCh, ldc,
                                                       M, N, K, alpha, relu, Hh);
    else
        gemm_bf16_wmma<false><<<grid, 256, 0, stream>>>(A, sAb, sAh, lda, Bm, sBb, sBh, ldb,
                                                        bias, C, Cbf, sCb, sCh, ldc,
                                                        M, N, K, alpha, relu, Hh);
}

static inline void launch_cvt(hipStream_t stream, const float* in, __bf16* out, int n)
{
    const int q = (n + 3) / 4;
    f32_to_bf16<<<(q + 255) / 256, 256, 0, stream>>>(in, out, n);
}

extern "C" void kernel_launch(void* const* d_in, const int* in_sizes, int n_in,
                              void* d_out, int out_size, void* d_ws, size_t ws_size,
                              hipStream_t stream)
{
    const float* enc        = (const float*)d_in[0];   // [B,S,D] (also cross-attn memory)
    const float* W_self_in  = (const float*)d_in[1];
    const float* b_self_in  = (const float*)d_in[2];
    const float* W_self_out = (const float*)d_in[3];
    const float* b_self_out = (const float*)d_in[4];
    const float* W_cross_in = (const float*)d_in[5];
    const float* b_cross_in = (const float*)d_in[6];
    const float* W_cross_out= (const float*)d_in[7];
    const float* b_cross_out= (const float*)d_in[8];
    const float* ln1_g = (const float*)d_in[9];
    const float* ln1_b = (const float*)d_in[10];
    const float* ln2_g = (const float*)d_in[11];
    const float* ln2_b = (const float*)d_in[12];
    const float* ln3_g = (const float*)d_in[13];
    const float* ln3_b = (const float*)d_in[14];
    const float* W_ff1 = (const float*)d_in[15];
    const float* b_ff1 = (const float*)d_in[16];
    const float* W_ff2 = (const float*)d_in[17];
    const float* b_ff2 = (const float*)d_in[18];
    const float* W_patch = (const float*)d_in[19];
    const float* b_patch = (const float*)d_in[20];

    const int M  = B_ * S_;        // 4096 tokens
    const int D3 = 3 * D_;         // 2304
    (void)in_sizes; (void)n_in; (void)ws_size;

    // ---------------- workspace layout ----------------
    // fp32 region
    float* ws     = (float*)d_ws;
    float* x      = ws;                                    // M*D
    float* scores = x + (size_t)M * D_;                    // B*H*S*S
    float* tmp    = scores + (size_t)B_ * H_ * S_ * S_;    // M*D
    float* ftmp   = scores;                                // final proj (aliases scores)
    // bf16 region
    __bf16* bfbase = (__bf16*)(tmp + (size_t)M * D_);
    __bf16* xbf    = bfbase;                               // M*D
    __bf16* qkvbf  = xbf    + (size_t)M * D_;              // M*3D
    __bf16* pbf    = qkvbf  + (size_t)M * D3;              // B*H*S*S (attn probs)
    __bf16* ctxbf  = pbf    + (size_t)B_ * H_ * S_ * S_;   // M*D
    __bf16* ff1bf  = pbf;                                  // M*DFF (aliases probs)
    // bf16 weights
    __bf16* wsi = ctxbf + (size_t)M * D_;                  // L*3D*D
    __bf16* wso = wsi + (size_t)L_ * D3 * D_;              // L*D*D
    __bf16* wci = wso + (size_t)L_ * D_ * D_;              // L*3D*D
    __bf16* wco = wci + (size_t)L_ * D3 * D_;              // L*D*D
    __bf16* wf1 = wco + (size_t)L_ * D_ * D_;              // L*DFF*D
    __bf16* wf2 = wf1 + (size_t)L_ * DFF_ * D_;            // L*D*DFF
    __bf16* wp  = wf2 + (size_t)L_ * D_ * DFF_;            // CPP*D
    __bf16* encbf = wp + (size_t)CPP_ * D_;                // M*D (bf16 memory for cross-attn)

    const size_t sQb = (size_t)S_ * D3;     // per-batch stride inside qkv
    const size_t sSb = (size_t)H_ * S_ * S_;
    const size_t sSh = (size_t)S_ * S_;
    const float  iscale = 0.125f;           // 1/sqrt(HD)

    // ---------------- one-time conversions ----------------
    launch_cvt(stream, W_self_in,  wsi, L_ * D3 * D_);
    launch_cvt(stream, W_self_out, wso, L_ * D_ * D_);
    launch_cvt(stream, W_cross_in, wci, L_ * D3 * D_);
    launch_cvt(stream, W_cross_out,wco, L_ * D_ * D_);
    launch_cvt(stream, W_ff1,      wf1, L_ * DFF_ * D_);
    launch_cvt(stream, W_ff2,      wf2, L_ * D_ * DFF_);
    launch_cvt(stream, W_patch,    wp,  CPP_ * D_);
    launch_cvt(stream, enc,        encbf, M * D_);

    // x = enc + PE
    {
        const int total = M * D_;
        add_pe<<<(total + 255) / 256, 256, 0, stream>>>(enc, x, xbf, total);
    }

    for (int l = 0; l < L_; ++l) {
        // ---------------- self-attention ----------------
        launch_gemm(stream, true, xbf, 0, 0, D_,
                    wsi + (size_t)l * D3 * D_, 0, 0, D_,
                    b_self_in + (size_t)l * D3,
                    nullptr, qkvbf, 0, 0, D3, M, D3, D_, 1.0f, 0, 1, 1);
        // scores = q k^T / sqrt(HD)
        launch_gemm(stream, true, qkvbf, sQb, HD_, D3,
                    qkvbf + D_, sQb, HD_, D3, nullptr,
                    scores, nullptr, sSb, sSh, S_, S_, S_, HD_, iscale, 0, H_, B_ * H_);
        softmax_rows<<<B_ * H_ * S_, 256, 0, stream>>>(scores, pbf, 1);
        // ctx = attn @ v
        launch_gemm(stream, false, pbf, sSb, sSh, S_,
                    qkvbf + 2 * D_, sQb, HD_, D3, nullptr,
                    nullptr, ctxbf, (size_t)S_ * D_, HD_, D_,
                    S_, HD_, S_, 1.0f, 0, H_, B_ * H_);
        // out-proj
        launch_gemm(stream, true, ctxbf, 0, 0, D_,
                    wso + (size_t)l * D_ * D_, 0, 0, D_,
                    b_self_out + (size_t)l * D_,
                    tmp, nullptr, 0, 0, D_, M, D_, D_, 1.0f, 0, 1, 1);
        add_layernorm<<<M, 256, 0, stream>>>(x, tmp, ln1_g + (size_t)l * D_,
                                             ln1_b + (size_t)l * D_, x, xbf, D_);

        // ---------------- cross-attention ----------------
        // q from x (cols 0..D-1 of qkv)
        launch_gemm(stream, true, xbf, 0, 0, D_,
                    wci + (size_t)l * D3 * D_, 0, 0, D_,
                    b_cross_in + (size_t)l * D3,
                    nullptr, qkvbf, 0, 0, D3, M, D_, D_, 1.0f, 0, 1, 1);
        // k,v from memory (cols D..3D-1 of qkv)
        launch_gemm(stream, true, encbf, 0, 0, D_,
                    wci + (size_t)l * D3 * D_ + (size_t)D_ * D_, 0, 0, D_,
                    b_cross_in + (size_t)l * D3 + D_,
                    nullptr, qkvbf + D_, 0, 0, D3, M, 2 * D_, D_, 1.0f, 0, 1, 1);
        launch_gemm(stream, true, qkvbf, sQb, HD_, D3,
                    qkvbf + D_, sQb, HD_, D3, nullptr,
                    scores, nullptr, sSb, sSh, S_, S_, S_, HD_, iscale, 0, H_, B_ * H_);
        softmax_rows<<<B_ * H_ * S_, 256, 0, stream>>>(scores, pbf, 0);
        launch_gemm(stream, false, pbf, sSb, sSh, S_,
                    qkvbf + 2 * D_, sQb, HD_, D3, nullptr,
                    nullptr, ctxbf, (size_t)S_ * D_, HD_, D_,
                    S_, HD_, S_, 1.0f, 0, H_, B_ * H_);
        launch_gemm(stream, true, ctxbf, 0, 0, D_,
                    wco + (size_t)l * D_ * D_, 0, 0, D_,
                    b_cross_out + (size_t)l * D_,
                    tmp, nullptr, 0, 0, D_, M, D_, D_, 1.0f, 0, 1, 1);
        add_layernorm<<<M, 256, 0, stream>>>(x, tmp, ln2_g + (size_t)l * D_,
                                             ln2_b + (size_t)l * D_, x, xbf, D_);

        // ---------------- FFN ----------------
        launch_gemm(stream, true, xbf, 0, 0, D_,
                    wf1 + (size_t)l * DFF_ * D_, 0, 0, D_,
                    b_ff1 + (size_t)l * DFF_,
                    nullptr, ff1bf, 0, 0, DFF_, M, DFF_, D_, 1.0f, 1, 1, 1);
        launch_gemm(stream, true, ff1bf, 0, 0, DFF_,
                    wf2 + (size_t)l * D_ * DFF_, 0, 0, DFF_,
                    b_ff2 + (size_t)l * D_,
                    tmp, nullptr, 0, 0, D_, M, D_, DFF_, 1.0f, 0, 1, 1);
        add_layernorm<<<M, 256, 0, stream>>>(x, tmp, ln3_g + (size_t)l * D_,
                                             ln3_b + (size_t)l * D_, x, xbf, D_);
    }

    // final projection + NHWC -> NCHW
    launch_gemm(stream, true, xbf, 0, 0, D_,
                wp, 0, 0, D_, b_patch,
                ftmp, nullptr, 0, 0, CPP_, M, CPP_, D_, 1.0f, 0, 1, 1);
    out_transpose<<<(out_size + 255) / 256, 256, 0, stream>>>(ftmp, (float*)d_out, out_size);
}
